// MaskInstanceLoss_51247549776210
// MI455X (gfx1250) — matrix operand order repaired and checked by reading, not compiled, not run on hardware
//
#include <hip/hip_runtime.h>
#include <hip/hip_bf16.h>
#include <math.h>

// ---------------------------------------------------------------------------
// NT-Xent-style masked instance loss, fused flash-LSE formulation for gfx1250.
//   B=4096, N=2B=8192, D=512, C=128, TEMP=0.5, pair(i) = i ^ 4096.
// Pass 0: split-convert z,c (fp32) -> bf16 hi/lo in workspace (~20 MB, L2-resident).
// Pass 1: 512 single-wave workgroups; each wave owns 16 rows, sweeps all 512
//         16-column tiles with V_WMMA_F32_16X16X32_BF16 (hi*hi + hi*lo + lo*hi
//         = ~fp32 accuracy). The 3 split terms use 3 INDEPENDENT accumulator
//         chains so the XDL pipe sees multi-way WMMA ILP instead of one serial
//         D->C dependency chain (important: ~430 VGPRs -> only ~2 waves/SIMD).
// Pass 2: deterministic tree reduction of 512 wave partials -> scalar loss.
// Workspace requirement: 20 MB + 2 KB.
// ---------------------------------------------------------------------------

typedef __attribute__((ext_vector_type(16))) __bf16 v16bf;
typedef __attribute__((ext_vector_type(8)))  __bf16 v8bf;
typedef __attribute__((ext_vector_type(8)))  float  v8f;

#define BATCH 4096
#define NROW  8192      // 2*BATCH
#define DDIM  512
#define CDIM  128
#define TEMPV 0.5f

// -------------------- Pass 0: fp32 -> bf16 hi/lo split ----------------------
__global__ __launch_bounds__(256)
void cvt_split_kernel(const float* __restrict__ zi, const float* __restrict__ zj,
                      const float* __restrict__ ci, const float* __restrict__ cj,
                      __bf16* __restrict__ zh, __bf16* __restrict__ zl,
                      __bf16* __restrict__ ch, __bf16* __restrict__ cl) {
    const int ZE = BATCH * DDIM;   // elements per z half
    const int CE = BATCH * CDIM;   // elements per c half
    int e = blockIdx.x * blockDim.x + threadIdx.x;
    if (e < ZE) {
        float a = zi[e];
        __bf16 h = (__bf16)a;
        zh[e] = h;
        zl[e] = (__bf16)(a - (float)h);
        float b = zj[e];
        h = (__bf16)b;
        zh[ZE + e] = h;
        zl[ZE + e] = (__bf16)(b - (float)h);
    }
    if (e < CE) {
        float a = ci[e];
        __bf16 h = (__bf16)a;
        ch[e] = h;
        cl[e] = (__bf16)(a - (float)h);
        float b = cj[e];
        h = (__bf16)b;
        ch[CE + e] = h;
        cl[CE + e] = (__bf16)(b - (float)h);
    }
}

// A-fragment load (16-bit A 16x32), contiguous K order matching B:
//   lane<16 : row=lane,     K = {k0..k0+15} with k0 = kc*32
//   lane>=16: row=lane-16,  K = {k0+16..k0+31}.
// rowPtr is already offset to this lane's row; kbase = kc*32 + (lane>=16 ? 16 : 0).
__device__ __forceinline__ v16bf load_a_frag(const __bf16* __restrict__ rowPtr, int kbase) {
    v8bf p0 = *(const v8bf*)(rowPtr + kbase);
    v8bf p1 = *(const v8bf*)(rowPtr + kbase + 8);
    v16bf r;
#pragma unroll
    for (int i = 0; i < 8; ++i) { r[i] = p0[i]; r[8 + i] = p1[i]; }
    return r;
}

#define WMMA_BF16(Afrag, Bfrag, Cacc) \
    __builtin_amdgcn_wmma_f32_16x16x32_bf16(false, (Afrag), false, (Bfrag), (short)0, (Cacc), false, false)

// -------------------- Pass 1: fused GEMM + online LSE -----------------------
__global__ __launch_bounds__(32)
void fused_loss_kernel(const __bf16* __restrict__ zh, const __bf16* __restrict__ zl,
                       const __bf16* __restrict__ ch, const __bf16* __restrict__ cl,
                       float* __restrict__ partial) {
    const int lane    = threadIdx.x;        // wave32
    const int rblk    = blockIdx.x;         // 512 blocks, 16 rows each
    const int half    = lane >> 4;          // 0: lanes 0-15, 1: lanes 16-31
    const int lane16  = lane & 15;
    const int aKoff   = half * 16;          // A-frag K sub-offset (contiguous 16)
    const int bKoff   = half * 16;          // B-frag K sub-offset (contiguous 16)

    const int rowA = rblk * 16 + lane16;    // this lane's A row
    const __bf16* zhRow = zh + (size_t)rowA * DDIM;
    const __bf16* zlRow = zl + (size_t)rowA * DDIM;
    const __bf16* chRow = ch + (size_t)rowA * CDIM;
    const __bf16* clRow = cl + (size_t)rowA * CDIM;

    // c is small (K=128): cache its A-fragments in VGPRs for the whole sweep.
    v16bf acH[4], acL[4];
#pragma unroll
    for (int kc = 0; kc < 4; ++kc) {
        acH[kc] = load_a_frag(chRow, kc * 32 + aKoff);
        acL[kc] = load_a_frag(clRow, kc * 32 + aKoff);
    }

    // Per-lane online-LSE state: VGPR slot v holds row rblk*16 + half*8 + v,
    // this lane's column subset (cols == lane16 mod 16).
    float m[8], l[8], pos[8];
#pragma unroll
    for (int v = 0; v < 8; ++v) { m[v] = -__builtin_inff(); l[v] = 0.f; pos[v] = 0.f; }

    const v8f vzero = {0.f,0.f,0.f,0.f,0.f,0.f,0.f,0.f};

    for (int jt = 0; jt < NROW / 16; ++jt) {
        const int colB = jt * 16 + lane16;  // B column owned by this lane (== C/D col N)
        const __bf16* bzH = zh + (size_t)colB * DDIM;
        const __bf16* bzL = zl + (size_t)colB * DDIM;
        const __bf16* bcH = ch + (size_t)colB * CDIM;
        const __bf16* bcL = cl + (size_t)colB * CDIM;

        // t-mask tile: c @ c^T  (K=128 -> 4 chunks x 3 split-terms = 12 WMMA)
        // Three independent accumulator chains -> multi-way XDL ILP.
        v8f c_hh = vzero, c_hl = vzero, c_lh = vzero;
#pragma unroll
        for (int kc = 0; kc < 4; ++kc) {
            v16bf bh = *(const v16bf*)(bcH + kc * 32 + bKoff);
            v16bf bl = *(const v16bf*)(bcL + kc * 32 + bKoff);
            c_hh = WMMA_BF16(acH[kc], bh, c_hh);
            c_hl = WMMA_BF16(acH[kc], bl, c_hl);
            c_lh = WMMA_BF16(acL[kc], bh, c_lh);
        }
        const v8f cacc = (c_hh + c_hl) + c_lh;

        // sim tile: z @ z^T  (K=512 -> 16 chunks x 3 split-terms = 48 WMMA)
        v8f z_hh = vzero, z_hl = vzero, z_lh = vzero;
#pragma unroll
        for (int kc = 0; kc < 16; ++kc) {
            v16bf ah = load_a_frag(zhRow, kc * 32 + aKoff);
            v16bf al = load_a_frag(zlRow, kc * 32 + aKoff);
            v16bf bh = *(const v16bf*)(bzH + kc * 32 + bKoff);
            v16bf bl = *(const v16bf*)(bzL + kc * 32 + bKoff);
            z_hh = WMMA_BF16(ah, bh, z_hh);
            z_hl = WMMA_BF16(ah, bl, z_hl);
            z_lh = WMMA_BF16(al, bh, z_lh);
        }
        const v8f zacc = (z_hh + z_hl) + z_lh;

        // Fold the 16x16 tile into the online LSE (C/D layout: VGPR v -> M=v+8*half).
        const int rbase = rblk * 16 + half * 8;
#pragma unroll
        for (int v = 0; v < 8; ++v) {
            const int row = rbase + v;
            float t = fmaxf(cacc[v], TEMPV);
            const bool isPair = (colB == (row ^ BATCH));   // (i+B) % N == i ^ 4096
            if (isPair) t = TEMPV;                          // forced temperature at positives
            // t >= 0.5 is well-conditioned: v_rcp_f32 + mul instead of full IEEE divide.
            const float s = zacc[v] * __builtin_amdgcn_rcpf(t);
            if (isPair) pos[v] += s;                        // exactly one hit per row
            if (colB != row) {                              // diagonal masked to -inf
                const float mn = fmaxf(m[v], s);
                l[v] = l[v] * __expf(m[v] - mn) + __expf(s - mn);
                m[v] = mn;
            }
        }
    }

    // Merge (m,l,pos) across the 16-lane half-groups (xor<16 stays inside a half).
#pragma unroll
    for (int off = 1; off < 16; off <<= 1) {
#pragma unroll
        for (int v = 0; v < 8; ++v) {
            const float m2 = __shfl_xor(m[v], off, 32);
            const float l2 = __shfl_xor(l[v], off, 32);
            const float p2 = __shfl_xor(pos[v], off, 32);
            const float mn = fmaxf(m[v], m2);
            l[v] = l[v] * __expf(m[v] - mn) + l2 * __expf(m2 - mn);
            m[v] = mn;
            pos[v] += p2;
        }
    }

    // Sum (lse - pos) over this half's 8 rows, then combine the two halves.
    float local = 0.f;
#pragma unroll
    for (int v = 0; v < 8; ++v) local += m[v] + __logf(l[v]) - pos[v];
    local += __shfl_xor(local, 16, 32);
    if (lane == 0) partial[rblk] = local;   // 16-row partial, all 512 blocks write
}

// -------------------- Pass 2: deterministic final reduction -----------------
__global__ __launch_bounds__(256)
void finalize_kernel(const float* __restrict__ partial, float* __restrict__ out) {
    __shared__ float smem[256];
    const int t = threadIdx.x;
    smem[t] = partial[t] + partial[t + 256];
    __syncthreads();
#pragma unroll
    for (int off = 128; off > 0; off >>= 1) {
        if (t < off) smem[t] += smem[t + off];
        __syncthreads();
    }
    if (t == 0) out[0] = smem[0] / (float)NROW;
}

// ---------------------------------------------------------------------------
extern "C" void kernel_launch(void* const* d_in, const int* in_sizes, int n_in,
                              void* d_out, int out_size, void* d_ws, size_t ws_size,
                              hipStream_t stream) {
    const float* zi = (const float*)d_in[0];   // [4096,512]
    const float* zj = (const float*)d_in[1];   // [4096,512]
    const float* ci = (const float*)d_in[2];   // [4096,128]
    const float* cj = (const float*)d_in[3];   // [4096,128]
    float* out = (float*)d_out;

    // Workspace carve (needs 20 MB + 2 KB):
    //   zh [8192*512] bf16 @ 0        (8 MB)
    //   zl [8192*512] bf16 @ 8 MB     (8 MB)
    //   ch [8192*128] bf16 @ 16 MB    (2 MB)
    //   cl [8192*128] bf16 @ 18 MB    (2 MB)
    //   partial [512] f32  @ 20 MB    (2 KB)
    char* ws = (char*)d_ws;
    const size_t ZB = (size_t)NROW * DDIM * sizeof(__bf16);   // 8 MB
    const size_t CB = (size_t)NROW * CDIM * sizeof(__bf16);   // 2 MB
    __bf16* zh = (__bf16*)(ws);
    __bf16* zl = (__bf16*)(ws + ZB);
    __bf16* ch = (__bf16*)(ws + 2 * ZB);
    __bf16* cl = (__bf16*)(ws + 2 * ZB + CB);
    float*  pr = (float*) (ws + 2 * ZB + 2 * CB);

    // Pass 0: split-convert (2,097,152 z elems; grid covers c's 524,288 too).
    cvt_split_kernel<<<(BATCH * DDIM) / 256, 256, 0, stream>>>(zi, zj, ci, cj, zh, zl, ch, cl);

    // Pass 1: 512 single-wave workgroups, 16 rows each.
    fused_loss_kernel<<<NROW / 16, 32, 0, stream>>>(zh, zl, ch, cl, pr);

    // Pass 2: scalar loss.
    finalize_kernel<<<1, 256, 0, stream>>>(pr, out);
}
